// SetOnlyLM_47098611368313
// MI455X (gfx1250) — matrix operand order repaired and checked by reading, not compiled, not run on hardware
//
#include <hip/hip_runtime.h>
#include <hip/hip_bf16.h>
#include <math.h>

// ---- model constants (must match reference) ----
#define D_MODEL 1024
#define SEQ_LEN 2048
#define BATCH   2
#define NSETS_  63
#define NHEAD   16
#define DHEAD   64
#define NLAYER  4
#define DFF_    4096
#define VOCAB   32000

typedef __attribute__((ext_vector_type(16))) __bf16          bf16x16;
typedef __attribute__((ext_vector_type(8)))  float           floatx8;
typedef __attribute__((ext_vector_type(8)))  unsigned short  ushort8;
typedef __attribute__((ext_vector_type(4)))  unsigned short  ushort4v;
typedef __attribute__((ext_vector_type(4)))  unsigned int    uint4v;
typedef __attribute__((ext_vector_type(8)))  unsigned int    uint8v;

__device__ __forceinline__ unsigned short f2bf(float f) {
  unsigned u = __float_as_uint(f);
  u += 0x7FFFu + ((u >> 16) & 1u);          // round-to-nearest-even
  return (unsigned short)(u >> 16);
}

union FragU { ushort8 u[2]; bf16x16 b; };

// ============================================================
// bf16 WMMA GEMM: C[M,N] = A[M,K] * B + (bias) (+gelu) (+resid)
//   TRANSB=0: B is [K,N] row-major (activation @ weight)
//   TRANSB=1: B is [N,K] row-major (activation @ weight^T)
// 256 threads (8 waves). Block tile 128x128, K-step 32.
// Wave w computes rows [16w,16w+16) x all 128 cols (8 v8f accums).
// ============================================================
template<bool TRANSB, bool GELU>
__global__ __launch_bounds__(256) void gemm_wmma(
    const float* __restrict__ A, const float* __restrict__ B,
    const float* __restrict__ bias, const float* __restrict__ resid,
    float* __restrict__ C, int M, int N, int K) {
  __shared__ unsigned short As[128 * 32];   // [m][k] bf16 bits
  __shared__ unsigned short Bs[128 * 32];   // [n][k] bf16 bits

  const int tid  = threadIdx.x;
  const int lane = tid & 31;
  const int wave = tid >> 5;
  const int half = lane >> 4;   // k-half selector (this code's contiguous k = 16*half + s)
  const int l16  = lane & 15;

  const int m_tile = blockIdx.y * 128;
  const int n_tile = blockIdx.x * 128;

  floatx8 acc[8];
  const floatx8 zero = {0.f,0.f,0.f,0.f,0.f,0.f,0.f,0.f};
#pragma unroll
  for (int j = 0; j < 8; ++j) acc[j] = zero;

  for (int kt = 0; kt < K; kt += 32) {
    __syncthreads();
    // ---- stage A tile (clamp rows for M=126 case) ----
    {
      const int kq = (tid & 7) * 4;
      const int mb = tid >> 3;
#pragma unroll
      for (int mi = 0; mi < 4; ++mi) {
        const int m = mb + 32 * mi;
        int gm = m_tile + m; if (gm > M - 1) gm = M - 1;
        const float4 av = *(const float4*)&A[(size_t)gm * K + kt + kq];
        ushort4v s = { f2bf(av.x), f2bf(av.y), f2bf(av.z), f2bf(av.w) };
        *(ushort4v*)&As[m * 32 + kq] = s;
        if (kt + 32 < K)
          __builtin_prefetch(&A[(size_t)gm * K + kt + 32 + kq], 0, 1);
      }
    }
    // ---- stage B tile ----
    if (TRANSB) {
      const int kq = (tid & 7) * 4;
      const int nb = tid >> 3;
#pragma unroll
      for (int ni = 0; ni < 4; ++ni) {
        const int n = nb + 32 * ni;
        const size_t gn = (size_t)(n_tile + n);
        const float4 bv = *(const float4*)&B[gn * K + kt + kq];
        ushort4v s = { f2bf(bv.x), f2bf(bv.y), f2bf(bv.z), f2bf(bv.w) };
        *(ushort4v*)&Bs[n * 32 + kq] = s;
        if (kt + 32 < K)
          __builtin_prefetch(&B[gn * K + kt + 32 + kq], 0, 1);
      }
    } else {
      const int n4 = (tid & 31) * 4;
      const int kb = tid >> 5;
#pragma unroll
      for (int ki = 0; ki < 4; ++ki) {
        const int k = kb + 8 * ki;
        const float4 bv = *(const float4*)&B[(size_t)(kt + k) * N + n_tile + n4];
        Bs[(n4 + 0) * 32 + k] = f2bf(bv.x);
        Bs[(n4 + 1) * 32 + k] = f2bf(bv.y);
        Bs[(n4 + 2) * 32 + k] = f2bf(bv.z);
        Bs[(n4 + 3) * 32 + k] = f2bf(bv.w);
      }
    }
    __syncthreads();

    // ---- A fragment: row = 16*wave + l16; lane half selects k runs
    FragU af;
    const int arow = wave * 16 + l16;
    af.u[0] = *(const ushort8*)&As[arow * 32 + 16 * half];
    af.u[1] = *(const ushort8*)&As[arow * 32 + 16 * half + 8];

    FragU barr[8];
#pragma unroll
    for (int j = 0; j < 8; ++j) {
      const int bcol = j * 16 + l16;
      barr[j].u[0] = *(const ushort8*)&Bs[bcol * 32 + 16 * half];
      barr[j].u[1] = *(const ushort8*)&Bs[bcol * 32 + 16 * half + 8];
    }
#pragma unroll
    for (int j = 0; j < 8; ++j)
      acc[j] = __builtin_amdgcn_wmma_f32_16x16x32_bf16(
          false, af.b, false, barr[j].b, (short)0, acc[j], false, false);
  }

  // ---- epilogue: C/D layout: VGPR r -> row (r + 8*half), col l16 ----
#pragma unroll
  for (int j = 0; j < 8; ++j) {
    const int col = n_tile + j * 16 + l16;
    const float bv = bias ? bias[col] : 0.f;
#pragma unroll
    for (int r = 0; r < 8; ++r) {
      const int row = m_tile + wave * 16 + half * 8 + r;
      if (row < M) {
        float x = acc[j][r] + bv;
        if (GELU) x = 0.5f * x * (1.f + erff(x * 0.70710678118654752f));
        if (resid) x += resid[(size_t)row * N + col];
        C[(size_t)row * N + col] = x;
      }
    }
  }
}

// ============================================================
// TDM-staged, double-buffered bf16 head GEMM: C = A[M,K] * B[N,K]^T
// Two LDS buffer pairs; wave 0 issues the next K-step's two D#s
// before waiting, then s_wait_tensorcnt<=2 (TDM completes in
// issue order) so DMA streams behind the WMMA chain.
// ============================================================
__device__ __forceinline__ void tdm_load_2d(
    unsigned lds_off, const unsigned short* gptr,
    unsigned tensor_d0, unsigned tensor_d1, unsigned stride0) {
  const unsigned long long ga = (unsigned long long)(uintptr_t)gptr;
  uint4v g0;
  g0[0] = 1u;                                                 // count=1, user D#
  g0[1] = lds_off;                                            // lds_addr (bytes)
  g0[2] = (unsigned)(ga & 0xFFFFFFFFull);                     // global_addr[31:0]
  g0[3] = (unsigned)((ga >> 32) & 0x1FFFFFFull) | (2u << 30); // addr[56:32] | type=2
  uint8v g1;
  g1[0] = 0x00010000u;                                        // data_size=1 (2B)
  g1[1] = (tensor_d0 & 0xFFFFu) << 16;                        // tensor_dim0[15:0]
  g1[2] = ((tensor_d0 >> 16) & 0xFFFFu) |
          ((tensor_d1 & 0xFFFFu) << 16);                      // dim0[31:16] | dim1[15:0]
  g1[3] = ((tensor_d1 >> 16) & 0xFFFFu) | (32u << 16);        // dim1[31:16] | tile_dim0=32
  g1[4] = 128u;                                               // tile_dim1=128, tile_dim2=0
  g1[5] = stride0;                                            // tensor_dim0_stride[31:0]
  g1[6] = 0u;
  g1[7] = 0u;
  asm volatile("tensor_load_to_lds %0, %1" :: "s"(g0), "s"(g1) : "memory");
}

__global__ __launch_bounds__(256) void gemm_head_tdm(
    const unsigned short* __restrict__ A,   // bf16 [M,K]
    const unsigned short* __restrict__ B,   // bf16 [N,K]
    float* __restrict__ C, int M, int N, int K) {
  __shared__ unsigned short As[2][128 * 32];
  __shared__ unsigned short Bs[2][128 * 32];

  const int tid  = threadIdx.x;
  const int lane = tid & 31;
  const int wave = tid >> 5;
  const int half = lane >> 4;
  const int l16  = lane & 15;
  const int m_tile = blockIdx.y * 128;
  const int n_tile = blockIdx.x * 128;

  const unsigned as_off0 = (unsigned)(uintptr_t)(void*)&As[0][0];
  const unsigned as_off1 = (unsigned)(uintptr_t)(void*)&As[1][0];
  const unsigned bs_off0 = (unsigned)(uintptr_t)(void*)&Bs[0][0];
  const unsigned bs_off1 = (unsigned)(uintptr_t)(void*)&Bs[1][0];

  const unsigned short* Abase = A + (size_t)m_tile * K;
  const unsigned short* Bbase = B + (size_t)n_tile * K;

  floatx8 acc[8];
  const floatx8 zero = {0.f,0.f,0.f,0.f,0.f,0.f,0.f,0.f};
#pragma unroll
  for (int j = 0; j < 8; ++j) acc[j] = zero;

  const int KSTEPS = K >> 5;

  // prologue: fill buffer 0
  if (wave == 0) {
    tdm_load_2d(as_off0, Abase, (unsigned)K, (unsigned)M, (unsigned)K);
    tdm_load_2d(bs_off0, Bbase, (unsigned)K, (unsigned)N, (unsigned)K);
  }

  for (int ks = 0; ks < KSTEPS; ++ks) {
    const int cur = ks & 1;
    // issue next K-step into the other buffer (all waves finished reading it
    // at the trailing barrier of iteration ks-1)
    if (ks + 1 < KSTEPS) {
      if (wave == 0) {
        const int kn = (ks + 1) << 5;
        tdm_load_2d(cur ? as_off0 : as_off1, Abase + kn, (unsigned)K, (unsigned)M, (unsigned)K);
        tdm_load_2d(cur ? bs_off0 : bs_off1, Bbase + kn, (unsigned)K, (unsigned)N, (unsigned)K);
      }
      __builtin_amdgcn_s_wait_tensorcnt(2);   // pair 'ks' landed (in-order TDM)
    } else {
      __builtin_amdgcn_s_wait_tensorcnt(0);   // last pair landed
    }
    __syncthreads();                          // LDS visible to all waves

    FragU af;
    const int arow = wave * 16 + l16;
    const unsigned short* as = &As[cur][0];
    const unsigned short* bs = &Bs[cur][0];
    af.u[0] = *(const ushort8*)&as[arow * 32 + 16 * half];
    af.u[1] = *(const ushort8*)&as[arow * 32 + 16 * half + 8];

    FragU barr[8];
#pragma unroll
    for (int j = 0; j < 8; ++j) {
      const int bcol = j * 16 + l16;
      barr[j].u[0] = *(const ushort8*)&bs[bcol * 32 + 16 * half];
      barr[j].u[1] = *(const ushort8*)&bs[bcol * 32 + 16 * half + 8];
    }
#pragma unroll
    for (int j = 0; j < 8; ++j)
      acc[j] = __builtin_amdgcn_wmma_f32_16x16x32_bf16(
          false, af.b, false, barr[j].b, (short)0, acc[j], false, false);

    __syncthreads();                          // done reading 'cur' before refill
  }

#pragma unroll
  for (int j = 0; j < 8; ++j) {
    const int col = n_tile + j * 16 + l16;
#pragma unroll
    for (int r = 0; r < 8; ++r) {
      const int row = m_tile + wave * 16 + half * 8 + r;
      if (row < M) C[(size_t)row * N + col] = acc[j][r];
    }
  }
}

// ============================================================
// elementwise / small kernels
// ============================================================
__global__ __launch_bounds__(256) void f2bf_kernel(
    const float* __restrict__ x, unsigned short* __restrict__ y) {
  const size_t gid = (size_t)blockIdx.x * 256 + threadIdx.x;
  y[gid] = f2bf(x[gid]);
}

__global__ __launch_bounds__(256) void embed_kernel(
    const int* __restrict__ ids, const float* __restrict__ tok_emb,
    const float* __restrict__ pos_emb, float* __restrict__ out) {
  const size_t gid = (size_t)blockIdx.x * 256 + threadIdx.x;   // < B*T*D
  const int d  = (int)(gid & (D_MODEL - 1));
  const int bt = (int)(gid >> 10);
  const int t  = bt & (SEQ_LEN - 1);
  const int id = ids[bt];
  out[gid] = tok_emb[(size_t)id * D_MODEL + d] + pos_emb[(size_t)t * D_MODEL + d];
}

// mean-pool 64 consecutive tokens (stride-32 windows) -> sets [B,S,D]
__global__ __launch_bounds__(256) void pool_kernel(
    const float* __restrict__ tok, float* __restrict__ sets) {
  const int gid = blockIdx.x * 256 + threadIdx.x;   // < B*NSETS*D
  const int d = gid & (D_MODEL - 1);
  const int s = (gid >> 10) % NSETS_;
  const int b = gid / (D_MODEL * NSETS_);
  const int t0 = s * 32;
  float acc = 0.f;
#pragma unroll 4
  for (int w = 0; w < 64; ++w)
    acc += tok[((size_t)(b * SEQ_LEN + t0 + w)) * D_MODEL + d];
  sets[gid] = acc * (1.f / 64.f);
}

// LayerNorm, one block per row of D=1024
__global__ __launch_bounds__(256) void ln_kernel(
    const float* __restrict__ x, const float* __restrict__ g,
    const float* __restrict__ bta, float* __restrict__ y) {
  __shared__ float r1[256], r2[256];
  const int row = blockIdx.x;
  const int tid = threadIdx.x;
  const float* xr = x + (size_t)row * D_MODEL;
  float s = 0.f, s2 = 0.f;
  for (int i = tid; i < D_MODEL; i += 256) { float v = xr[i]; s += v; s2 += v * v; }
  r1[tid] = s; r2[tid] = s2; __syncthreads();
  for (int off = 128; off > 0; off >>= 1) {
    if (tid < off) { r1[tid] += r1[tid + off]; r2[tid] += r2[tid + off]; }
    __syncthreads();
  }
  const float mean = r1[0] * (1.f / D_MODEL);
  const float var  = r2[0] * (1.f / D_MODEL) - mean * mean;
  const float inv  = rsqrtf(var + 1e-5f);
  for (int i = tid; i < D_MODEL; i += 256)
    y[(size_t)row * D_MODEL + i] = (xr[i] - mean) * inv * g[i] + bta[i];
}

// banded attention over sets: one 64-thread block per (b,h,s)
__global__ __launch_bounds__(64) void attn_kernel(
    const float* __restrict__ q, const float* __restrict__ k,
    const float* __restrict__ v, float* __restrict__ o) {
  __shared__ float red[64];
  __shared__ float sc[9];
  const int idx = blockIdx.x;                 // b*H*S
  const int s = idx % NSETS_;
  const int h = (idx / NSETS_) % NHEAD;
  const int b = idx / (NSETS_ * NHEAD);
  const int dh = threadIdx.x;
  const size_t rowq = ((size_t)(b * NSETS_ + s)) * D_MODEL + h * DHEAD;
  const float qd = q[rowq + dh];
  const int t0 = (s - 4 > 0) ? s - 4 : 0;
  const int t1 = (s + 4 < NSETS_ - 1) ? s + 4 : NSETS_ - 1;
  for (int t = t0; t <= t1; ++t) {
    const size_t rowk = ((size_t)(b * NSETS_ + t)) * D_MODEL + h * DHEAD;
    red[dh] = qd * k[rowk + dh];
    __syncthreads();
    for (int off = 32; off > 0; off >>= 1) {
      if (dh < off) red[dh] += red[dh + off];
      __syncthreads();
    }
    if (dh == 0) sc[t - t0] = red[0] * 0.125f;   // 1/sqrt(64)
    __syncthreads();
  }
  const int nt = t1 - t0 + 1;
  float mx = -1e30f;
  for (int j = 0; j < nt; ++j) mx = fmaxf(mx, sc[j]);
  float w[9], den = 0.f;
  for (int j = 0; j < nt; ++j) { w[j] = __expf(sc[j] - mx); den += w[j]; }
  float accv = 0.f;
  for (int j = 0; j < nt; ++j) {
    const size_t rowv = ((size_t)(b * NSETS_ + t0 + j)) * D_MODEL + h * DHEAD;
    accv += w[j] * v[rowv + dh];
  }
  o[rowq + dh] = accv / den;
}

// uniform router + residual: h[b,t,d] = tok + mean(sets covering t)
__global__ __launch_bounds__(256) void router_kernel(
    const float* __restrict__ tok, const float* __restrict__ sets,
    float* __restrict__ h) {
  const size_t gid = (size_t)blockIdx.x * 256 + threadIdx.x;   // < B*T*D
  const int d = (int)(gid & (D_MODEL - 1));
  const int t = (int)((gid >> 10) & (SEQ_LEN - 1));
  const int b = (int)(gid >> 21);
  const int a = t - 63;
  const int s_lo = (a <= 0) ? 0 : ((a + 31) >> 5);
  int s_hi = t >> 5; if (s_hi > NSETS_ - 1) s_hi = NSETS_ - 1;
  float acc = 0.f;
  for (int s = s_lo; s <= s_hi; ++s)
    acc += sets[((size_t)(b * NSETS_ + s)) * D_MODEL + d];
  h[gid] = tok[gid] + acc / (float)(s_hi - s_lo + 1);
}

// ============================================================
// launcher
// ============================================================
extern "C" void kernel_launch(void* const* d_in, const int* in_sizes, int n_in,
                              void* d_out, int out_size, void* d_ws, size_t ws_size,
                              hipStream_t stream) {
  const int*   ids     = (const int*)  d_in[0];
  const float* tok_emb = (const float*)d_in[1];
  const float* pos_emb = (const float*)d_in[2];
  const float* mlp_w1  = (const float*)d_in[3];
  const float* mlp_b1  = (const float*)d_in[4];
  const float* mlp_w2  = (const float*)d_in[5];
  const float* mlp_b2  = (const float*)d_in[6];
  const float* ln1_g   = (const float*)d_in[7];
  const float* ln1_b   = (const float*)d_in[8];
  const float* wq      = (const float*)d_in[9];
  const float* wk      = (const float*)d_in[10];
  const float* wv      = (const float*)d_in[11];
  const float* wo      = (const float*)d_in[12];
  const float* ln2_g   = (const float*)d_in[13];
  const float* ln2_b   = (const float*)d_in[14];
  const float* ff_w1   = (const float*)d_in[15];
  const float* ff_b1   = (const float*)d_in[16];
  const float* ff_w2   = (const float*)d_in[17];
  const float* ff_b2   = (const float*)d_in[18];
  const float* head_w  = (const float*)d_in[19];
  float* out = (float*)d_out;

  const size_t TOKN = (size_t)BATCH * SEQ_LEN * D_MODEL;   // 4,194,304
  const size_t SETN = (size_t)BATCH * NSETS_ * D_MODEL;    // 129,024
  const size_t HWN  = (size_t)VOCAB * D_MODEL;             // 32,768,000
  float* ws     = (float*)d_ws;
  float* tokpre = ws;
  float* tokmid = ws + TOKN;
  float* tok    = ws + 2 * TOKN;
  float* hbuf   = ws + 3 * TOKN;
  float* sets   = ws + 4 * TOKN;
  float* lbuf   = sets + SETN;
  float* qb     = lbuf + SETN;
  float* kb     = qb + SETN;
  float* vb     = kb + SETN;
  float* ob     = vb + SETN;
  float* ffb    = ob + SETN;                       // [B*S, DFF]
  unsigned short* hbuf_bf  = (unsigned short*)(ffb + (size_t)BATCH * NSETS_ * DFF_);
  unsigned short* headw_bf = hbuf_bf + TOKN;       // bf16 head_w [V,D]

  const int MTOK = BATCH * SEQ_LEN;   // 4096
  const int MSET = BATCH * NSETS_;    // 126

  // 1) embed
  embed_kernel<<<(unsigned)(TOKN / 256), 256, 0, stream>>>(ids, tok_emb, pos_emb, tokpre);

  // 2) token MLP (two D x D GEMMs, first with exact GELU)
  gemm_wmma<false, true ><<<dim3(D_MODEL / 128, MTOK / 128), 256, 0, stream>>>(
      tokpre, mlp_w1, mlp_b1, nullptr, tokmid, MTOK, D_MODEL, D_MODEL);
  gemm_wmma<false, false><<<dim3(D_MODEL / 128, MTOK / 128), 256, 0, stream>>>(
      tokmid, mlp_w2, mlp_b2, nullptr, tok, MTOK, D_MODEL, D_MODEL);

  // 3) window mean-pool -> sets
  pool_kernel<<<(unsigned)(SETN / 256), 256, 0, stream>>>(tok, sets);

  // 4) set-transformer layers
  for (int l = 0; l < NLAYER; ++l) {
    const size_t wOff  = (size_t)l * D_MODEL * D_MODEL;
    const size_t f1Off = (size_t)l * D_MODEL * DFF_;
    const size_t f2Off = (size_t)l * DFF_ * D_MODEL;
    ln_kernel<<<MSET, 256, 0, stream>>>(sets, ln1_g + l * D_MODEL, ln1_b + l * D_MODEL, lbuf);
    gemm_wmma<false, false><<<dim3(D_MODEL / 128, 1), 256, 0, stream>>>(
        lbuf, wq + wOff, nullptr, nullptr, qb, MSET, D_MODEL, D_MODEL);
    gemm_wmma<false, false><<<dim3(D_MODEL / 128, 1), 256, 0, stream>>>(
        lbuf, wk + wOff, nullptr, nullptr, kb, MSET, D_MODEL, D_MODEL);
    gemm_wmma<false, false><<<dim3(D_MODEL / 128, 1), 256, 0, stream>>>(
        lbuf, wv + wOff, nullptr, nullptr, vb, MSET, D_MODEL, D_MODEL);
    attn_kernel<<<BATCH * NHEAD * NSETS_, 64, 0, stream>>>(qb, kb, vb, ob);
    gemm_wmma<false, false><<<dim3(D_MODEL / 128, 1), 256, 0, stream>>>(
        ob, wo + wOff, nullptr, sets, sets, MSET, D_MODEL, D_MODEL);
    ln_kernel<<<MSET, 256, 0, stream>>>(sets, ln2_g + l * D_MODEL, ln2_b + l * D_MODEL, lbuf);
    gemm_wmma<false, true ><<<dim3(DFF_ / 128, 1), 256, 0, stream>>>(
        lbuf, ff_w1 + f1Off, ff_b1 + (size_t)l * DFF_, nullptr, ffb, MSET, DFF_, D_MODEL);
    gemm_wmma<false, false><<<dim3(D_MODEL / 128, 1), 256, 0, stream>>>(
        ffb, ff_w2 + f2Off, ff_b2 + (size_t)l * D_MODEL, sets, sets, MSET, D_MODEL, DFF_);
  }

  // 5) router + residual
  router_kernel<<<(unsigned)(TOKN / 256), 256, 0, stream>>>(tok, sets, hbuf);

  // 6) bf16 conversions for the TDM head GEMM
  f2bf_kernel<<<(unsigned)(TOKN / 256), 256, 0, stream>>>(hbuf, hbuf_bf);
  f2bf_kernel<<<(unsigned)(HWN / 256), 256, 0, stream>>>(head_w, headw_bf);

  // 7) head GEMM via double-buffered TDM staging (bulk of FLOPs)
  gemm_head_tdm<<<dim3(VOCAB / 128, MTOK / 128), 256, 0, stream>>>(
      hbuf_bf, headw_bf, out, MTOK, VOCAB, D_MODEL);
}
